// SelfAttention_60206851556087
// MI455X (gfx1250) — matrix operand order repaired and checked by reading, not compiled, not run on hardware
//
#include <hip/hip_runtime.h>

#define B_ 4
#define S_ 4096
#define D_ 768
#define KT 32            // WMMA K depth per step (f16)
#define NK (D_ / KT)     // 24 k-steps over D
#define BR 64            // queries per block (attention)
#define BC 64            // keys per tile (attention)

typedef _Float16 h16;
typedef __attribute__((ext_vector_type(16))) _Float16 v16h;
typedef __attribute__((ext_vector_type(8)))  _Float16 v8h;
typedef __attribute__((ext_vector_type(8)))  float    v8f;
typedef __attribute__((ext_vector_type(4)))  float    v4f;

__device__ inline v16h join8(v8h lo, v8h hi) {
  v16h r;
  #pragma unroll
  for (int i = 0; i < 8; ++i) { r[i] = lo[i]; r[8 + i] = hi[i]; }
  return r;
}

// A fragment (16x32, f16 row-major, stride ld halves). Lane&15 = row M.
// Lanes 0-15 hold K {0..7,16..23}; lanes 16-31 hold K {8..15,24..31}.
__device__ inline v16h a_frag_h(const h16* p, int ld, int lane) {
  int r = lane & 15, ko = (lane >> 4) << 3;
  const h16* q = p + (size_t)r * ld + ko;
  v8h lo = *(const v8h*)q;
  v8h hi = *(const v8h*)(q + 16);
  return join8(lo, hi);
}

// B fragment (32x16, f16): column n's K-data contiguous at p + n*ld.
// Lanes 0-15 hold K {0..7,16..23}; lanes 16-31 hold K {8..15,24..31}.
__device__ inline v16h b_frag_h(const h16* p, int ld, int lane) {
  int n = lane & 15, ko = (lane >> 4) << 3;
  const h16* q = p + (size_t)n * ld + ko;
  v8h lo = *(const v8h*)q;
  v8h hi = *(const v8h*)(q + 16);
  return join8(lo, hi);
}

// A fragment with on-the-fly f32 -> f16 conversion (row-major f32, stride ld).
__device__ inline v16h a_frag_f32(const float* p, int ld, int lane) {
  int r = lane & 15, ko = (lane >> 4) << 3;
  const float* q = p + (size_t)r * ld + ko;
  v4f a0 = *(const v4f*)q;
  v4f a1 = *(const v4f*)(q + 4);
  v4f b0 = *(const v4f*)(q + 16);
  v4f b1 = *(const v4f*)(q + 20);
  v16h o;
  #pragma unroll
  for (int i = 0; i < 4; ++i) {
    o[i]      = (h16)a0[i];
    o[4 + i]  = (h16)a1[i];
    o[8 + i]  = (h16)b0[i];
    o[12 + i] = (h16)b1[i];
  }
  return o;
}

// B fragment with on-the-fly f32 -> f16 conversion (column n at p + n*ld).
__device__ inline v16h b_frag_f32(const float* p, int ld, int lane) {
  int n = lane & 15, ko = (lane >> 4) << 3;
  const float* q = p + (size_t)n * ld + ko;
  v4f c0 = *(const v4f*)q;
  v4f c1 = *(const v4f*)(q + 4);
  v4f c2 = *(const v4f*)(q + 16);
  v4f c3 = *(const v4f*)(q + 20);
  v16h o;
  #pragma unroll
  for (int i = 0; i < 4; ++i) {
    o[i]      = (h16)c0[i];
    o[4 + i]  = (h16)c1[i];
    o[8 + i]  = (h16)c2[i];
    o[12 + i] = (h16)c3[i];
  }
  return o;
}

// CDNA5 async global -> LDS copy, 16 bytes per lane (GV addressing mode).
// ldsb = workgroup-relative LDS byte address, gsrc = 64-bit global address.
__device__ inline void async_copy_b128(unsigned int ldsb, const h16* gsrc) {
  asm volatile("global_load_async_to_lds_b128 %0, %1, off"
               :: "v"(ldsb), "v"(gsrc) : "memory");
}
__device__ inline void wait_asynccnt0() {
  asm volatile("s_wait_asynccnt 0x0" ::: "memory");
}

// ---------------------------------------------------------------------------
// Kernel 1: QKV projection. y = x @ W^T for W in {Wq,Wk,Wv}.
// Block = 256 threads (8 waves) owns one 16-row tile of x; the 8 waves share
// the 3*48 = 144 (matrix, 16-col-tile) jobs. Q,K stored f16 row-major;
// V stored f16 transposed per batch: Vt[b][e][s].
// ---------------------------------------------------------------------------
__global__ __launch_bounds__(256) void qkv_kernel(
    const float* __restrict__ x,
    const float* __restrict__ Wq, const float* __restrict__ Wk,
    const float* __restrict__ Wv,
    h16* __restrict__ Qh, h16* __restrict__ Kh, h16* __restrict__ Vt) {
  int s0 = blockIdx.x * 16;           // row in flattened [B*S, D]
  int b  = s0 / S_;
  int sb = s0 - b * S_;               // row within batch
  int w = threadIdx.x >> 5, lane = threadIdx.x & 31;
  int col = lane & 15, rbase = (lane >> 4) << 3;

  for (int j = w; j < 3 * (D_ / 16); j += 8) {
    int m  = j / (D_ / 16);
    int n0 = (j - m * (D_ / 16)) * 16;
    const float* Wm = (m == 0) ? Wq : ((m == 1) ? Wk : Wv);

    v8f c = {};
    #pragma unroll
    for (int ks = 0; ks < NK; ++ks) {
      v16h a  = a_frag_f32(x + (size_t)s0 * D_ + ks * KT, D_, lane);
      v16h bf = b_frag_f32(Wm + (size_t)n0 * D_ + ks * KT, D_, lane);
      c = __builtin_amdgcn_wmma_f32_16x16x32_f16(false, a, false, bf,
                                                 (short)0, c, false, false);
    }
    if (m == 2) {
      #pragma unroll
      for (int r = 0; r < 8; ++r) {
        int srow = sb + rbase + r;
        Vt[((size_t)b * D_ + n0 + col) * S_ + srow] = (h16)c[r];
      }
    } else {
      h16* dst = (m == 0) ? Qh : Kh;
      #pragma unroll
      for (int r = 0; r < 8; ++r)
        dst[((size_t)s0 + rbase + r) * D_ + n0 + col] = (h16)c[r];
    }
  }
}

// ---------------------------------------------------------------------------
// Kernel 2: causal flash attention. Block = 256 threads (8 waves), 64 queries,
// 64-key tiles. Per tile:
//   Stage:   async-copy the contiguous 96KB K tile global->LDS
//            (global_load_async_to_lds_b128 + s_wait_asynccnt).
//   Phase A: wave (qsub = w>>1, ksel = w&1) computes a 16x32 score strip
//            (48 WMMAs over D=768, K B-frags from LDS), scales, masks -> LDS.
//   Phase B: threads 0..63 run online softmax per row (m, l, alpha in LDS),
//            write P as f16 to LDS.
//   Phase C: wave (qsub = w>>1, fh = w&1) rescales its 24 f32 accumulators and
//            does 48 P.V WMMAs (P A-frags from LDS, V B-frags from Vt).
// ---------------------------------------------------------------------------
__global__ __launch_bounds__(256) void attn_kernel(
    const h16* __restrict__ Qh, const h16* __restrict__ Kh,
    const h16* __restrict__ Vt, float* __restrict__ out) {
  __shared__ __align__(16) h16 sK[BC * D_];     // 96 KB staged K tile
  __shared__ float sS[BR * BC];                  // 16 KB scores
  __shared__ __align__(16) h16 sP[BR * BC];     // 8 KB probabilities
  __shared__ float sM[BR], sL[BR], sA[BR];

  int b  = blockIdx.x / (S_ / BR);
  int qt = blockIdx.x % (S_ / BR);
  int q0 = qt * BR;
  const h16* Qb = Qh + (size_t)b * S_ * D_;
  const h16* Kb = Kh + (size_t)b * S_ * D_;
  const h16* Vb = Vt + (size_t)b * D_ * S_;
  float* Ob = out + (size_t)b * S_ * D_;

  int tid = threadIdx.x, w = tid >> 5, lane = tid & 31;
  int qsub = w >> 1;          // 0..3 : 16-query sub-tile
  int ksel = w & 1;           // score phase: which 32-key half of the tile
  int fh   = w & 1;           // PV phase: which 384-feature half
  int col = lane & 15, rbase = (lane >> 4) << 3;

  if (tid < BR) { sM[tid] = -__builtin_inff(); sL[tid] = 0.f; }

  v8f o[24];
  {
    v8f z = {};
    #pragma unroll
    for (int t = 0; t < 24; ++t) o[t] = z;
  }
  __syncthreads();

  const int nkt = qt + 1;                     // 64-key tiles covering 0..q0+63
  const float sc = 0.03608439182435161f;      // 1/sqrt(768)
  const unsigned int sKb = (unsigned int)(size_t)&sK[0];

  for (int kt = 0; kt < nkt; ++kt) {
    // ---- Stage K tile (contiguous 96KB) into LDS via async copy ----
    {
      const h16* ksrc = Kb + (size_t)kt * BC * D_;
      #pragma unroll 4
      for (int i = tid; i < (BC * D_) / 8; i += 256)
        async_copy_b128(sKb + (unsigned int)i * 16u, ksrc + (size_t)i * 8);
      wait_asynccnt0();
    }
    // warm L2 for next iteration's K tile while this one is consumed
    if (kt + 1 < nkt) {
      const h16* knext = Kb + (size_t)(kt + 1) * BC * D_;
      #pragma unroll
      for (int i = 0; i < 3; ++i)
        __builtin_prefetch(knext + (size_t)tid * 192 + i * 64, 0, 0);
    }
    __syncthreads();

    // ---- Phase A: 16x32 score strip per wave (K from LDS) ----
    v8f s0 = {}, s1 = {};
    const h16* Qp = Qb + (size_t)(q0 + qsub * 16) * D_;
    const h16* Kp0 = sK + (size_t)(ksel * 32) * D_;
    const h16* Kp1 = sK + (size_t)(ksel * 32 + 16) * D_;
    #pragma unroll
    for (int ks = 0; ks < NK; ++ks) {
      v16h a   = a_frag_h(Qp + ks * KT, D_, lane);
      v16h bk0 = b_frag_h(Kp0 + ks * KT, D_, lane);
      v16h bk1 = b_frag_h(Kp1 + ks * KT, D_, lane);
      s0 = __builtin_amdgcn_wmma_f32_16x16x32_f16(false, a, false, bk0,
                                                  (short)0, s0, false, false);
      s1 = __builtin_amdgcn_wmma_f32_16x16x32_f16(false, a, false, bk1,
                                                  (short)0, s1, false, false);
    }
    #pragma unroll
    for (int r = 0; r < 8; ++r) {
      int row = qsub * 16 + rbase + r;        // block-local query row
      int c0 = ksel * 32 + col;               // tile-local key columns
      int c1 = ksel * 32 + 16 + col;
      float v0 = s0[r] * sc;
      float v1 = s1[r] * sc;
      if (kt * BC + c0 > q0 + row) v0 = -__builtin_inff();
      if (kt * BC + c1 > q0 + row) v1 = -__builtin_inff();
      sS[row * BC + c0] = v0;
      sS[row * BC + c1] = v1;
    }
    __syncthreads();

    // ---- Phase B: online softmax over the 64-key tile ----
    if (tid < BR) {
      int row = tid;
      float mold = sM[row];
      float mrow = mold;
      #pragma unroll
      for (int c2 = 0; c2 < BC; ++c2) mrow = fmaxf(mrow, sS[row * BC + c2]);
      float alpha = __expf(mold - mrow);
      float rs = 0.f;
      #pragma unroll
      for (int c2 = 0; c2 < BC; ++c2) {
        float p = __expf(sS[row * BC + c2] - mrow);
        sP[row * BC + c2] = (h16)p;
        rs += p;
      }
      sM[row] = mrow;
      sL[row] = sL[row] * alpha + rs;
      sA[row] = alpha;
    }
    __syncthreads();

    // ---- Phase C: rescale accumulators, P @ V (two 32-key chunks) ----
    float al[8];
    #pragma unroll
    for (int r = 0; r < 8; ++r) al[r] = sA[qsub * 16 + rbase + r];
    v16h aP0 = a_frag_h(sP + qsub * 16 * BC, BC, lane);
    v16h aP1 = a_frag_h(sP + qsub * 16 * BC + 32, BC, lane);
    #pragma unroll
    for (int t = 0; t < 24; ++t) {
      int f0 = fh * 384 + t * 16;
      const h16* vp = Vb + (size_t)f0 * S_ + (size_t)kt * BC;
      v16h bv0 = b_frag_h(vp, S_, lane);
      v16h bv1 = b_frag_h(vp + 32, S_, lane);
      if (kt + 1 < nkt) __builtin_prefetch(vp + BC, 0, 0);
      #pragma unroll
      for (int r = 0; r < 8; ++r) o[t][r] *= al[r];
      o[t] = __builtin_amdgcn_wmma_f32_16x16x32_f16(false, aP0, false, bv0,
                                                    (short)0, o[t], false, false);
      o[t] = __builtin_amdgcn_wmma_f32_16x16x32_f16(false, aP1, false, bv1,
                                                    (short)0, o[t], false, false);
    }
    __syncthreads();
  }

  // ---- Epilogue: divide by l, store f32 ----
  float linv[8];
  #pragma unroll
  for (int r = 0; r < 8; ++r) linv[r] = 1.f / sL[qsub * 16 + rbase + r];
  #pragma unroll
  for (int t = 0; t < 24; ++t) {
    int f0 = fh * 384 + t * 16;
    #pragma unroll
    for (int r = 0; r < 8; ++r) {
      int row = q0 + qsub * 16 + rbase + r;
      Ob[(size_t)row * D_ + f0 + col] = o[t][r] * linv[r];
    }
  }
}

extern "C" void kernel_launch(void* const* d_in, const int* in_sizes, int n_in,
                              void* d_out, int out_size, void* d_ws, size_t ws_size,
                              hipStream_t stream) {
  (void)in_sizes; (void)n_in; (void)out_size; (void)ws_size;
  const float* x  = (const float*)d_in[0];
  const float* Wq = (const float*)d_in[1];
  const float* Wk = (const float*)d_in[2];
  const float* Wv = (const float*)d_in[3];
  float* out = (float*)d_out;

  // Workspace layout (f16): Qh | Kh | Vt  (3 * B*S*D * 2 bytes = 75.5 MB)
  h16* Qh = (h16*)d_ws;
  h16* Kh = Qh + (size_t)B_ * S_ * D_;
  h16* Vt = Kh + (size_t)B_ * S_ * D_;

  qkv_kernel<<<dim3((B_ * S_) / 16), 256, 0, stream>>>(x, Wq, Wk, Wv, Qh, Kh, Vt);
  attn_kernel<<<dim3(B_ * (S_ / BR)), 256, 0, stream>>>(Qh, Kh, Vt, out);
}
